// UNet_52372831207681
// MI455X (gfx1250) — matrix operand order repaired and checked by reading, not compiled, not run on hardware
//
#include <hip/hip_runtime.h>
#include <hip/hip_bf16.h>
#include <math.h>

typedef __attribute__((ext_vector_type(16))) _Float16 v16h;
typedef __attribute__((ext_vector_type(8)))  _Float16 v8h;
typedef __attribute__((ext_vector_type(8)))  float    v8f;
typedef __attribute__((ext_vector_type(4)))  float    v4f;

#define BB    4      // batch
#define NPTS  4096
#define NPAR  1024
#define KNNK  16
#define DC    128    // DCTX

// ---------------------------------------------------------------------------
// Generic fused GEMM:  C[M,N] = act( (A[M,K] @ W[K,N] + bias) * g + be )
// f16 inputs, f32 WMMA accumulation (v_wmma_f32_16x16x32_f16), wave32.
//
// Block: 128 threads = 4 waves. Tile BM=128 x BN=64, K-step 32.
// Wave w computes rows [32w, 32w+32) -> 2 row-frags x 4 col-frags = 8 WMMA.
//
// LDS tiles are FRAGMENT-MAJOR (CDNA5 ISA 7.12.2 wave32 layout), so a
// fragment load is one contiguous 32-byte read per lane (2x ds_load_b128).
// Staging fast path (K multiple of 32): each lane segment of the A tile is
// two contiguous K-octets -> 4x global b128 + 2x ds_store_b128; B segments
// pack 16 coalesced column loads into two v8h stores. K=3 inputs use the
// scalar fallback path.
// ---------------------------------------------------------------------------
#define FRAG_H 40   // halves per lane segment (80 B: 16B-aligned, bank-spread)

__global__ __launch_bounds__(128) void unet_gemm_kernel(
    const float* __restrict__ A, const float* __restrict__ W,
    const float* __restrict__ bias, const float* __restrict__ g,
    const float* __restrict__ be, float* __restrict__ C,
    int M, int N, int Kd, int act)
{
  __shared__ _Float16 Af[8 * 32 * FRAG_H];  // 8 row-frags x 32 lanes x 16h (padded)
  __shared__ _Float16 Bf[4 * 32 * FRAG_H];  // 4 col-frags x 32 lanes x 16h (padded)

  const int tid  = threadIdx.x;
  const int wave = tid >> 5;
  const int lane = tid & 31;
  const int m0   = blockIdx.x * 128;
  const int n0   = blockIdx.y * 64;

  v8f acc[2][4] = {};

  for (int k0 = 0; k0 < Kd; k0 += 32) {
    const bool kfull = (Kd - k0) >= 32;

    if (kfull) {
      // ---- fast A stage: 2 segments per thread, vector loads/stores ----
#pragma unroll
      for (int s = 0; s < 2; ++s) {
        int seg = tid + s * 128;            // seg = frag*32 + lane
        int ls  = seg & 31;
        int m   = m0 + ((seg >> 5) << 4) + (ls & 15);
        int kb  = k0 + ((ls >> 4) << 3);    // lanes16-31 -> K base +8
        v8h hlo = {}, hhi = {};
        if (m < M) {
          const float* ap = A + (size_t)m * Kd + kb;
          v4f x0 = *(const v4f*)ap;
          v4f x1 = *(const v4f*)(ap + 4);
          v4f x2 = *(const v4f*)(ap + 16);
          v4f x3 = *(const v4f*)(ap + 20);
#pragma unroll
          for (int e = 0; e < 4; ++e) {
            hlo[e]     = (_Float16)x0[e];
            hlo[4 + e] = (_Float16)x1[e];
            hhi[e]     = (_Float16)x2[e];
            hhi[4 + e] = (_Float16)x3[e];
          }
        }
        *(v8h*)&Af[seg * FRAG_H]     = hlo;
        *(v8h*)&Af[seg * FRAG_H + 8] = hhi;
      }
      // ---- fast B stage: 1 segment per thread, coalesced column loads ----
      {
        int ls = tid & 31;
        int n  = n0 + ((tid >> 5) << 4) + (ls & 15);
        int kb = k0 + ((ls >> 4) << 3);     // lanes16-31 -> K base +8
        v8h hlo = {}, hhi = {};
        if (n < N) {
#pragma unroll
          for (int e = 0; e < 8; ++e) hlo[e] = (_Float16)W[(size_t)(kb + e) * N + n];
#pragma unroll
          for (int e = 0; e < 8; ++e) hhi[e] = (_Float16)W[(size_t)(kb + 16 + e) * N + n];
        }
        *(v8h*)&Bf[tid * FRAG_H]     = hlo;
        *(v8h*)&Bf[tid * FRAG_H + 8] = hhi;
      }
    } else {
      // ---- scalar fallback (K=3 inputs): zero-padded per-element scatter ----
      for (int i = tid; i < 128 * 32; i += 128) {
        int r = i >> 5, k = i & 31;
        int gm = m0 + r, gk = k0 + k;
        float v = (gm < M && gk < Kd) ? A[(size_t)gm * Kd + gk] : 0.f;
        int hi = (k >> 3) & 1;
        int e  = (k & 7) + ((k >> 4) << 3);
        int ln = (hi << 4) + (r & 15);
        Af[((r >> 4) * 32 + ln) * FRAG_H + e] = (_Float16)v;
      }
      for (int i = tid; i < 32 * 64; i += 128) {
        int k = i >> 6, n = i & 63;
        int gk = k0 + k, gn = n0 + n;
        float v = (gk < Kd && gn < N) ? W[(size_t)gk * N + gn] : 0.f;
        int ln = (((k >> 3) & 1) << 4) + (n & 15);
        Bf[((n >> 4) * 32 + ln) * FRAG_H + ((k & 7) + ((k >> 4) << 3))] = (_Float16)v;
      }
    }
    __syncthreads();

    v16h a0 = *(const v16h*)&Af[(((wave << 1) + 0) * 32 + lane) * FRAG_H];
    v16h a1 = *(const v16h*)&Af[(((wave << 1) + 1) * 32 + lane) * FRAG_H];
#pragma unroll
    for (int t = 0; t < 4; ++t) {
      v16h bf = *(const v16h*)&Bf[(t * 32 + lane) * FRAG_H];
      acc[0][t] = __builtin_amdgcn_wmma_f32_16x16x32_f16(false, a0, false, bf,
                                                         (short)0, acc[0][t], false, false);
      acc[1][t] = __builtin_amdgcn_wmma_f32_16x16x32_f16(false, a1, false, bf,
                                                         (short)0, acc[1][t], false, false);
    }
    __syncthreads();
  }

  // epilogue: C layout: VGPR i -> M = i (lanes 0-15) / i+8 (lanes 16-31); N = lane&15
  const int ncolL = lane & 15;
  const int mhalf = (lane >> 4) << 3;
#pragma unroll
  for (int rf = 0; rf < 2; ++rf) {
    const int mbase = m0 + (wave << 5) + (rf << 4) + mhalf;
#pragma unroll
    for (int t = 0; t < 4; ++t) {
      int n = n0 + t * 16 + ncolL;
      if (n >= N) continue;
      float bv  = bias ? bias[n] : 0.f;
      float gv  = g ? g[n] : 1.f;
      float bev = g ? be[n] : 0.f;
#pragma unroll
      for (int i = 0; i < 8; ++i) {
        int m = mbase + i;
        if (m >= M) continue;
        float v = (acc[rf][t][i] + bv) * gv + bev;
        if (act == 1)      v = fmaxf(v, 0.f);
        else if (act == 2) v = v / (1.f + __expf(-v));   // silu
        C[(size_t)m * N + n] = v;
      }
    }
  }
}

// ---------------------------------------------------------------------------
// brute-force kNN with LDS point tiling + register insertion sort
// ---------------------------------------------------------------------------
template <int KQ>
__global__ __launch_bounds__(256) void unet_knn_kernel(
    const float* __restrict__ srcp, const float* __restrict__ dstp,
    int* __restrict__ outidx, int Ns, int Nd, int excl, int appendself, int kout)
{
  __shared__ float sx[256], sy[256], sz[256];
  const int b = blockIdx.y;
  const int q = blockIdx.x * 256 + threadIdx.x;
  const bool valid = q < Nd;
  float qx = 0.f, qy = 0.f, qz = 0.f;
  if (valid) {
    const float* p = dstp + ((size_t)b * Nd + q) * 3;
    qx = p[0]; qy = p[1]; qz = p[2];
  }
  float bd[KQ]; int bi[KQ];
#pragma unroll
  for (int j = 0; j < KQ; ++j) { bd[j] = 3.4e38f; bi[j] = 0; }

  for (int s0 = 0; s0 < Ns; s0 += 256) {
    int cnt = Ns - s0; if (cnt > 256) cnt = 256;
    __syncthreads();
    if (threadIdx.x < cnt) {
      const float* p = srcp + ((size_t)b * Ns + s0 + threadIdx.x) * 3;
      sx[threadIdx.x] = p[0]; sy[threadIdx.x] = p[1]; sz[threadIdx.x] = p[2];
    }
    __syncthreads();
    if (!valid) continue;
    for (int j = 0; j < cnt; ++j) {
      int s = s0 + j;
      if (excl && s == q) continue;
      float dx = qx - sx[j], dy = qy - sy[j], dz = qz - sz[j];
      float d2 = dx * dx + dy * dy + dz * dz;
      if (d2 < bd[KQ - 1]) {
        bd[KQ - 1] = d2; bi[KQ - 1] = s;
#pragma unroll
        for (int t = KQ - 1; t > 0; --t) {
          if (bd[t] < bd[t - 1]) {
            float tv = bd[t]; bd[t] = bd[t - 1]; bd[t - 1] = tv;
            int   ti = bi[t]; bi[t] = bi[t - 1]; bi[t - 1] = ti;
          }
        }
      }
    }
  }
  if (valid) {
    int* o = outidx + ((size_t)b * Nd + q) * kout;
#pragma unroll
    for (int j = 0; j < KQ; ++j) o[j] = bi[j];
    if (appendself) o[kout - 1] = q;
  }
}

// ---------------------------------------------------------------------------
// farthest point sampling: one block per cloud, d[] in LDS, block argmax
// ---------------------------------------------------------------------------
__global__ __launch_bounds__(1024) void unet_fps_kernel(
    const float* __restrict__ pos, int* __restrict__ outidx, int Np, int m)
{
  __shared__ float d[4096];
  __shared__ float rv[1024];
  __shared__ int   ri[1024];
  const int b = blockIdx.x, tid = threadIdx.x;
  const float* P = pos + (size_t)b * Np * 3;
  float p0x = P[0], p0y = P[1], p0z = P[2];
  for (int i = tid; i < Np; i += 1024) {
    float dx = P[i * 3] - p0x, dy = P[i * 3 + 1] - p0y, dz = P[i * 3 + 2] - p0z;
    d[i] = dx * dx + dy * dy + dz * dz;
  }
  if (tid == 0) outidx[(size_t)b * m] = 0;
  __syncthreads();
  for (int it = 1; it < m; ++it) {
    float lv = -1.f; int li = 0;
    for (int i = tid; i < Np; i += 1024)
      if (d[i] > lv) { lv = d[i]; li = i; }
    rv[tid] = lv; ri[tid] = li;
    __syncthreads();
    for (int s = 512; s > 0; s >>= 1) {
      if (tid < s) {
        float v2 = rv[tid + s]; int i2 = ri[tid + s];
        if (v2 > rv[tid] || (v2 == rv[tid] && i2 < ri[tid])) { rv[tid] = v2; ri[tid] = i2; }
      }
      __syncthreads();
    }
    int nxt = ri[0];
    if (tid == 0) outidx[(size_t)b * m + it] = nxt;
    float px = P[nxt * 3], py = P[nxt * 3 + 1], pz = P[nxt * 3 + 2];
    for (int i = tid; i < Np; i += 1024) {
      float dx = P[i * 3] - px, dy = P[i * 3 + 1] - py, dz = P[i * 3 + 2] - pz;
      float nd = dx * dx + dy * dy + dz * dz;
      if (nd < d[i]) d[i] = nd;
    }
    __syncthreads();
  }
}

__global__ void unet_gather_pos_kernel(const float* __restrict__ pos,
                                       const int* __restrict__ idx,
                                       float* __restrict__ out, int Np, int m)
{
  int t = blockIdx.x * blockDim.x + threadIdx.x;
  if (t >= BB * m) return;
  int b = t / m;
  const float* p = pos + ((size_t)b * Np + idx[t]) * 3;
  float* o = out + (size_t)t * 3;
  o[0] = p[0]; o[1] = p[1]; o[2] = p[2];
}

__global__ void unet_gather_pd_kernel(const float* __restrict__ pdst,
                                      const float* __restrict__ psrc,
                                      const int* __restrict__ nbr,
                                      float* __restrict__ pd, int Nd, int Ns, int k)
{
  int idx = blockIdx.x * blockDim.x + threadIdx.x;   // (b*Nd+i)*k + j
  if (idx >= BB * Nd * k) return;
  int bi = idx / k;
  int b  = bi / Nd;
  int s  = nbr[idx];
  const float* pq = pdst + (size_t)bi * 3;
  const float* ps = psrc + ((size_t)b * Ns + s) * 3;
  float* o = pd + (size_t)idx * 3;
  o[0] = pq[0] - ps[0]; o[1] = pq[1] - ps[1]; o[2] = pq[2] - ps[2];
}

__global__ void unet_combine_e_kernel(const float* __restrict__ adst,
                                      const float* __restrict__ asrc,
                                      const float* __restrict__ delta,
                                      const int* __restrict__ nbr,
                                      float* __restrict__ e,
                                      int Nd, int Ns, int k, int C)
{
  size_t idx = (size_t)blockIdx.x * blockDim.x + threadIdx.x;
  size_t total = (size_t)BB * Nd * k * C;
  if (idx >= total) return;
  int c = (int)(idx % C);
  size_t r = idx / C;              // (b*Nd+i)*k + j
  size_t bi = r / k;
  int b = (int)(bi / Nd);
  int s = nbr[r];
  e[idx] = adst[bi * C + c] - asrc[((size_t)b * Ns + s) * C + c] + delta[idx];
}

__global__ void unet_ptconv_out_kernel(const float* __restrict__ alpha,
                                       const float* __restrict__ lin,
                                       const float* __restrict__ delta,
                                       const int* __restrict__ nbr,
                                       float* __restrict__ out,
                                       int Nd, int Ns, int k, int C)
{
  size_t idx = (size_t)blockIdx.x * blockDim.x + threadIdx.x;  // (b*Nd+i)*C + c
  if (idx >= (size_t)BB * Nd * C) return;
  int c = (int)(idx % C);
  size_t bi = idx / C;
  int b = (int)(bi / Nd);
  size_t base = bi * k * C + c;
  const int* nb = nbr + bi * k;
  float mx = -3.4e38f;
  for (int j = 0; j < k; ++j) mx = fmaxf(mx, alpha[base + (size_t)j * C]);
  float se = 0.f, acc = 0.f;
  for (int j = 0; j < k; ++j) {
    float w = __expf(alpha[base + (size_t)j * C] - mx);
    int s = nb[j];
    acc += w * (lin[((size_t)b * Ns + s) * C + c] + delta[base + (size_t)j * C]);
    se += w;
  }
  out[idx] = acc / se;
}

__global__ void unet_scatter_mean_kernel(const float* __restrict__ hm,
                                         const int* __restrict__ nbr,
                                         float* __restrict__ out,
                                         int Nd, int Ns, int C)
{
  size_t idx = (size_t)blockIdx.x * blockDim.x + threadIdx.x;  // (b*Nd+i)*C + c
  if (idx >= (size_t)BB * Nd * C) return;
  int c = (int)(idx % C);
  size_t bi = idx / C;
  int b = (int)(bi / Nd);
  const int* nb = nbr + bi * KNNK;
  float s = 0.f;
  for (int j = 0; j < KNNK; ++j) s += hm[((size_t)b * Ns + nb[j]) * C + c];
  out[idx] = s * (1.f / (float)KNNK);
}

__global__ void unet_interp_add_kernel(const float* __restrict__ xs,
                                       const float* __restrict__ psub,
                                       const float* __restrict__ pup,
                                       const int* __restrict__ nbr,
                                       float* __restrict__ out,
                                       int Nsub, int Nup, int C)
{
  size_t idx = (size_t)blockIdx.x * blockDim.x + threadIdx.x;  // (b*Nup+i)*C + c
  if (idx >= (size_t)BB * Nup * C) return;
  int c = (int)(idx % C);
  size_t bi = idx / C;
  int b = (int)(bi / Nup);
  const float* pu = pup + bi * 3;
  const int* nb = nbr + bi * 3;
  float ws = 0.f, acc = 0.f;
  for (int j = 0; j < 3; ++j) {
    int s = nb[j];
    const float* ps = psub + ((size_t)b * Nsub + s) * 3;
    float dx = pu[0] - ps[0], dy = pu[1] - ps[1], dz = pu[2] - ps[2];
    float w = 1.f / fmaxf(dx * dx + dy * dy + dz * dz, 1e-16f);
    ws += w;
    acc += w * xs[((size_t)b * Nsub + s) * C + c];
  }
  out[idx] += acc / ws;
}

__global__ void unet_time_embed_kernel(const float* __restrict__ t, float* __restrict__ ctx)
{
  int i = blockIdx.x * blockDim.x + threadIdx.x;   // b*128 + d
  if (i >= BB * DC) return;
  int b = i / DC, d = i % DC;
  int half = DC / 2;
  int f = (d < half) ? d : d - half;
  float freq = __expf(-__logf(10000.f) * (float)f / (float)half);
  float a = t[b] * freq;
  ctx[b * (2 * DC) + d] = (d < half) ? sinf(a) : cosf(a);
}

__global__ void unet_maxreduce_kernel(const float* __restrict__ he2, float* __restrict__ ctx)
{
  int i = blockIdx.x * blockDim.x + threadIdx.x;   // b*128 + c
  if (i >= BB * DC) return;
  int b = i / DC, c = i % DC;
  float m = -3.4e38f;
  for (int p = 0; p < NPAR; ++p) m = fmaxf(m, he2[((size_t)b * NPAR + p) * DC + c]);
  ctx[b * (2 * DC) + DC + c] = m;
}

// tiny M=B matvec for ctx projections (act: 0 none, 1 relu, 2 silu, 3 sigmoid)
__global__ void unet_ctxproj_kernel(const float* __restrict__ x, const float* __restrict__ W,
                                    const float* __restrict__ bias, const float* __restrict__ g,
                                    const float* __restrict__ be, float* __restrict__ out,
                                    int N, int Kd, int act)
{
  int i = blockIdx.x * blockDim.x + threadIdx.x;   // b*N + n
  if (i >= BB * N) return;
  int b = i / N, n = i % N;
  const float* xr = x + (size_t)b * Kd;
  float s = bias ? bias[n] : 0.f;
  for (int k = 0; k < Kd; ++k) s += xr[k] * W[(size_t)k * N + n];
  if (g) s = s * g[n] + be[n];
  if (act == 1)      s = fmaxf(s, 0.f);
  else if (act == 2) s = s / (1.f + __expf(-s));
  else if (act == 3) s = 1.f / (1.f + __expf(-s));
  out[i] = s;
}

__global__ void unet_bcast_add_kernel(float* __restrict__ h, const float* __restrict__ cp,
                                      int Np, int C)
{
  size_t idx = (size_t)blockIdx.x * blockDim.x + threadIdx.x;
  if (idx >= (size_t)BB * Np * C) return;
  int c = (int)(idx % C);
  int b = (int)(idx / ((size_t)Np * C));
  h[idx] += cp[b * C + c];
}

__global__ void unet_final_kernel(const float* __restrict__ relu3, const float* __restrict__ lin3,
                                  const float* __restrict__ gate, const float* __restrict__ biasv,
                                  float* __restrict__ out)
{
  int idx = blockIdx.x * blockDim.x + threadIdx.x;   // (b*N+i)*3 + c
  if (idx >= BB * NPTS * 3) return;
  int c = idx % 3;
  int b = idx / (NPTS * 3);
  out[idx] = relu3[idx] + lin3[idx] * gate[b * 3 + c] + biasv[b * 3 + c];
}

// ---------------------------------------------------------------------------
// host orchestration
// ---------------------------------------------------------------------------
struct Layer { const float *W, *b, *g, *be; };
struct Conv  { const float *Wdst, *Wlin, *Wsrc; };
struct TBp   { Layer attn0, attn1; Conv conv; Layer mlp_in, mlp_out, pos0, pos1; };
struct CMp   { Layer ctx, mlp; };

extern "C" void kernel_launch(void* const* d_in, const int* in_sizes, int n_in,
                              void* d_out, int out_size, void* d_ws, size_t ws_size,
                              hipStream_t stream) {
  (void)in_sizes; (void)n_in; (void)out_size; (void)ws_size;

  const float* pos = (const float*)d_in[0];   // [B*N, 3]
  const float* tin = (const float*)d_in[1];   // [B]
  const float* par = (const float*)d_in[2];   // [B*NP, 3]

  // ---- parameter leaves in jax tree_flatten (alphabetical-dict) order ----
  int pi = 5;
  auto nx = [&]() -> const float* { return (const float*)d_in[pi++]; };
  auto rdL = [&](bool norm) {
    Layer L; L.W = nx(); L.b = nx();
    if (norm) { L.be = nx(); L.g = nx(); } else { L.be = nullptr; L.g = nullptr; }
    return L;
  };
  auto rdConv = [&]() { Conv c; c.Wdst = nx(); c.Wlin = nx(); c.Wsrc = nx(); return c; };
  auto rdTB = [&]() {
    TBp t;
    t.attn0 = rdL(false); t.attn1 = rdL(false);
    t.conv = rdConv();
    t.mlp_in = rdL(true); t.mlp_out = rdL(true);
    t.pos0 = rdL(false); t.pos1 = rdL(false);
    return t;
  };
  auto rdCM = [&]() { CMp c; c.ctx = rdL(true); c.mlp = rdL(true); return c; };

  CMp down_cm[2]; Layer down_mlp[2]; TBp down_tb[2];
  for (int i = 0; i < 2; ++i) { down_cm[i] = rdCM(); down_mlp[i] = rdL(true); down_tb[i] = rdTB(); }
  CMp emb_cm = rdCM();
  const float* encW1 = nx(); const float* encW2 = nx();
  const float* encb1 = nx(); const float* encb2 = nx();
  const float* gWb = nx(); const float* gWg = nx(); const float* gWl = nx();
  const float* gbg = nx(); const float* gbl = nx();
  Layer mlp_input = rdL(true);
  Layer mlp_out_l = rdL(false);
  Layer mlp_par   = rdL(true);
  Layer mlp_summit = rdL(false); (void)mlp_summit;          // dead branch
  Layer pt_attn = rdL(false); Conv pt_cv = rdConv(); Layer pt_pos = rdL(false);
  TBp tb_input  = rdTB();
  TBp tb_summit = rdTB(); (void)tb_summit;                   // dead branch
  CMp up_cm[2]; Layer up_mlp[2], up_mlp_sub[2]; TBp up_tb[2];
  for (int i = 0; i < 2; ++i) {
    up_cm[i] = rdCM(); up_mlp[i] = rdL(true); up_mlp_sub[i] = rdL(true); up_tb[i] = rdTB();
  }
  // down[1], up[1], summit are dead (their results are overwritten/discarded
  // by the reference forward) -- only the live path is computed below.

  // ---- workspace carve-up ----
  float* wp = (float*)d_ws;
  size_t off = 0;
  auto alloc = [&](size_t n) { float* p = wp + off; off += (n + 63) & ~(size_t)63; return p; };
  const size_t BIG  = (size_t)BB * 4096 * 17 * 64;   // == 4*2048*17*128
  const size_t FBUF = (size_t)BB * 4096 * 128;

  float* tbig1 = alloc(BIG);
  float* tbig2 = alloc(BIG);
  float* tbig3 = alloc(BIG);
  float* linb  = alloc(FBUF);
  float* asrc  = alloc(FBUF);
  float* adst  = alloc(FBUF);
  float* tb1   = alloc(FBUF);
  float* tb2   = alloc(FBUF);
  float* hA    = alloc(FBUF);
  float* hB    = alloc(FBUF);
  float* hC    = alloc(FBUF);
  float* h0    = alloc(FBUF);
  float* h1    = alloc(FBUF);
  float* henc  = alloc((size_t)BB * NPAR * DC);
  float* he2   = alloc((size_t)BB * NPAR * DC);
  float* pd    = alloc((size_t)BB * 4096 * 17 * 3);
  float* ctxb  = alloc(BB * 256);
  float* cproj = alloc(BB * 256);
  float* gate  = alloc(16);
  float* biasv = alloc(16);
  float* lin3  = alloc((size_t)BB * NPTS * 3);
  float* relu3 = alloc((size_t)BB * NPTS * 3);
  float* pos1  = alloc((size_t)BB * 2048 * 3);
  int* nbr    = (int*)alloc((size_t)BB * 4096 * 17);
  int* nbr3   = (int*)alloc((size_t)BB * 4096 * 3);
  int* fpsidx = (int*)alloc((size_t)BB * 2048);

  auto gemm = [&](const float* A, const Layer& L, float* C, int M, int N, int Kd, int act) {
    dim3 grid((M + 127) / 128, (N + 63) / 64);
    unet_gemm_kernel<<<grid, 128, 0, stream>>>(A, L.W, L.b, L.g, L.be, C, M, N, Kd, act);
  };
  auto gemmW = [&](const float* A, const float* W, const float* bias, float* C,
                   int M, int N, int Kd, int act) {
    dim3 grid((M + 127) / 128, (N + 63) / 64);
    unet_gemm_kernel<<<grid, 128, 0, stream>>>(A, W, bias, nullptr, nullptr, C, M, N, Kd, act);
  };
  auto L1D = [&](size_t total) { return dim3((unsigned)((total + 255) / 256)); };

  // PointTransformerConv
  auto run_ptconv = [&](const float* xsrc, const float* xdst, const float* psrc,
                        const float* pdst, const int* nbrp, int Ns, int Nd, int k, int C,
                        const Conv& cv, const Layer* posL, int nPos,
                        const Layer* attnL, int nAttn, float* outp) {
    int Msrc = BB * Ns, Mdst = BB * Nd;
    size_t Me = (size_t)BB * Nd * k;
    gemmW(xsrc, cv.Wlin, nullptr, linb, Msrc, C, C, 0);
    gemmW(xsrc, cv.Wsrc, nullptr, asrc, Msrc, C, C, 0);
    gemmW(xdst, cv.Wdst, nullptr, adst, Mdst, C, C, 0);
    unet_gather_pd_kernel<<<L1D(Me), 256, 0, stream>>>(pdst, psrc, nbrp, pd, Nd, Ns, k);
    if (nPos == 2) {
      gemm(pd, posL[0], tbig3, (int)Me, C, 3, 2);
      gemm(tbig3, posL[1], tbig1, (int)Me, C, C, 2);
    } else {
      gemm(pd, posL[0], tbig1, (int)Me, C, 3, 2);
    }
    unet_combine_e_kernel<<<L1D(Me * C), 256, 0, stream>>>(adst, asrc, tbig1, nbrp, tbig2,
                                                          Nd, Ns, k, C);
    const float* alpha;
    if (nAttn == 2) {
      gemm(tbig2, attnL[0], tbig3, (int)Me, C, C, 2);
      gemm(tbig3, attnL[1], tbig2, (int)Me, C, C, 2);
      alpha = tbig2;
    } else {
      gemm(tbig2, attnL[0], tbig3, (int)Me, C, C, 2);
      alpha = tbig3;
    }
    unet_ptconv_out_kernel<<<L1D((size_t)BB * Nd * C), 256, 0, stream>>>(
        alpha, linb, tbig1, nbrp, outp, Nd, Ns, k, C);
  };

  // TransformerBlock
  auto run_tb = [&](const TBp& T, const float* hin, float* outp, const float* posp,
                    int Np, int C) {
    dim3 kg((Np + 255) / 256, BB);
    unet_knn_kernel<16><<<kg, 256, 0, stream>>>(posp, posp, nbr, Np, Np, 1, 1, 17);
    gemm(hin, T.mlp_in, tb1, BB * Np, C, C, 2);
    Layer posL[2]  = {T.pos0, T.pos1};
    Layer attnL[2] = {T.attn0, T.attn1};
    run_ptconv(tb1, tb1, posp, posp, nbr, Np, Np, 17, C, T.conv, posL, 2, attnL, 2, tb2);
    gemm(tb2, T.mlp_out, outp, BB * Np, C, C, 0);
  };

  // ConditioningMerge
  auto run_cm = [&](const CMp& Cm, const float* hin, float* outp, int Np, int C) {
    gemm(hin, Cm.mlp, outp, BB * Np, C, C, 2);
    unet_ctxproj_kernel<<<L1D(BB * C), 256, 0, stream>>>(ctxb, Cm.ctx.W, Cm.ctx.b,
                                                        Cm.ctx.g, Cm.ctx.be, cproj, C, 256, 2);
    unet_bcast_add_kernel<<<L1D((size_t)BB * Np * C), 256, 0, stream>>>(outp, cproj, Np, C);
  };

  // ===== live forward path =====
  // context: time embed + PointNet global max
  unet_time_embed_kernel<<<L1D(BB * DC), 256, 0, stream>>>(tin, ctxb);
  gemmW(par, encW1, encb1, henc, BB * NPAR, DC, 3, 2);      // silu
  gemmW(henc, encW2, encb2, he2, BB * NPAR, DC, DC, 0);
  unet_maxreduce_kernel<<<L1D(BB * DC), 256, 0, stream>>>(he2, ctxb);

  // input feature MLPs
  gemm(pos, mlp_input, hA, BB * NPTS, 64, 3, 1);            // relu
  gemm(par, mlp_par,   hB, BB * NPAR, 64, 3, 1);            // relu

  // FeatureTransfer par -> pos (k=16, single-layer pos/attn nns)
  {
    dim3 kg((NPTS + 255) / 256, BB);
    unet_knn_kernel<16><<<kg, 256, 0, stream>>>(par, pos, nbr, NPAR, NPTS, 0, 0, 16);
    run_ptconv(hB, hA, par, pos, nbr, NPAR, NPTS, 16, 64, pt_cv, &pt_pos, 1, &pt_attn, 1, hC);
  }
  run_cm(emb_cm, hC, hA, NPTS, 64);
  run_tb(tb_input, hA, h0, pos, NPTS, 64);                  // skip level 0

  // down[0]: 4096 -> 2048, 64 -> 128
  unet_fps_kernel<<<BB, 1024, 0, stream>>>(pos, fpsidx, NPTS, 2048);
  unet_gather_pos_kernel<<<L1D((size_t)BB * 2048), 256, 0, stream>>>(pos, fpsidx, pos1, NPTS, 2048);
  {
    dim3 kg((2048 + 255) / 256, BB);
    unet_knn_kernel<16><<<kg, 256, 0, stream>>>(pos, pos1, nbr, NPTS, 2048, 0, 0, 16);
  }
  gemm(h0, down_mlp[0], hA, BB * NPTS, 128, 64, 2);         // silu
  unet_scatter_mean_kernel<<<L1D((size_t)BB * 2048 * 128), 256, 0, stream>>>(
      hA, nbr, hB, 2048, NPTS, 128);
  run_tb(down_tb[0], hB, hC, pos1, 2048, 128);
  run_cm(down_cm[0], hC, h1, 2048, 128);                    // skip level 1

  // up (only the last iteration is live): h1 -> interp onto h0's resolution
  gemm(h1, up_mlp_sub[0], hA, BB * 2048, 64, 128, 1);       // xs, relu
  gemm(h0, up_mlp[0],     hB, BB * NPTS, 64, 64, 1);        // base, relu
  {
    dim3 kg((NPTS + 255) / 256, BB);
    unet_knn_kernel<3><<<kg, 256, 0, stream>>>(pos1, pos, nbr3, 2048, NPTS, 0, 0, 3);
  }
  unet_interp_add_kernel<<<L1D((size_t)BB * NPTS * 64), 256, 0, stream>>>(
      hA, pos1, pos, nbr3, hB, 2048, NPTS, 64);
  run_tb(up_tb[0], hB, hC, pos, NPTS, 64);
  run_cm(up_cm[0], hC, hA, NPTS, 64);                       // hA = final features

  // GLU head + residual relu MLP
  gemmW(hA, gWl, gbl, lin3, BB * NPTS, 3, 64, 0);
  gemmW(hA, mlp_out_l.W, mlp_out_l.b, relu3, BB * NPTS, 3, 64, 1);
  unet_ctxproj_kernel<<<1, 256, 0, stream>>>(ctxb, gWg, gbg, nullptr, nullptr, gate, 3, 256, 3);
  unet_ctxproj_kernel<<<1, 256, 0, stream>>>(ctxb, gWb, nullptr, nullptr, nullptr, biasv, 3, 256, 0);
  unet_final_kernel<<<L1D((size_t)BB * NPTS * 3), 256, 0, stream>>>(
      relu3, lin3, gate, biasv, (float*)d_out);
}
